// BP_35132832481311
// MI455X (gfx1250) — compile-verified, matrix operand order checked by reading, not run on hardware
//
#include <hip/hip_runtime.h>
#include <hip/hip_bf16.h>
#include <math.h>

#define Q 8
#define NUM_ITER 10

typedef __attribute__((ext_vector_type(16))) __bf16 v16bf;
typedef __attribute__((ext_vector_type(8)))  float  v8f;

// ---------- helpers ----------

static __device__ __forceinline__ unsigned short f32_to_bf16(float f) {
    union { float f; unsigned int u; } x; x.f = f;
    unsigned int r = x.u + 0x7FFFu + ((x.u >> 16) & 1u);   // round-to-nearest-even
    return (unsigned short)(r >> 16);
}

static __device__ __forceinline__ void load_row8(const float* __restrict__ p, float v[Q]) {
    const float4* r = reinterpret_cast<const float4*>(p);
    float4 a = r[0], b = r[1];
    v[0]=a.x; v[1]=a.y; v[2]=a.z; v[3]=a.w;
    v[4]=b.x; v[5]=b.y; v[6]=b.z; v[7]=b.w;
}

static __device__ __forceinline__ void store_row8(float* __restrict__ p, const float v[Q]) {
    float4 a = make_float4(v[0],v[1],v[2],v[3]);
    float4 b = make_float4(v[4],v[5],v[6],v[7]);
    float4* r = reinterpret_cast<float4*>(p);
    r[0]=a; r[1]=b;
}

static __device__ __forceinline__ void softmax8(float v[Q]) {
    float m = v[0];
    #pragma unroll
    for (int q = 1; q < Q; ++q) m = fmaxf(m, v[q]);
    float s = 0.f;
    #pragma unroll
    for (int q = 0; q < Q; ++q) { v[q] = __expf(v[q] - m); s += v[q]; }
    float inv = 1.0f / s;
    #pragma unroll
    for (int q = 0; q < Q; ++q) v[q] *= inv;
}

// ---------- kernel 0: psi = softmax(psi0) ----------

__global__ void k_init_psi(const float* __restrict__ psi0, float* __restrict__ psi, int twoE) {
    int e = blockIdx.x * blockDim.x + threadIdx.x;
    if (e >= twoE) return;
    float v[Q];
    load_row8(psi0 + (size_t)e * Q, v);
    softmax8(v);
    store_row8(psi + (size_t)e * Q, v);
}

// ---------- kernel 1: log_term + segment-sum scatter into node_sum ----------
// src_dir[e] = ei[e]; dst_dir[e] = ei[rev(e)], rev(e) = e<E ? e+E : e-E.

__global__ void k_scatter(const float* __restrict__ psi, const int* __restrict__ ei,
                          const float* __restrict__ beta,
                          float* __restrict__ log_term, float* __restrict__ node_sum,
                          int E, int twoE) {
    int e = blockIdx.x * blockDim.x + threadIdx.x;
    if (e >= twoE) return;
    float w = __expf(beta[0]) - 1.0f;
    __builtin_prefetch(psi + (size_t)(e + 4096) * Q, 0, 0);  // global_prefetch_b8, speculative
    float v[Q];
    load_row8(psi + (size_t)e * Q, v);
    #pragma unroll
    for (int q = 0; q < Q; ++q) v[q] = log1pf(w * v[q]);
    store_row8(log_term + (size_t)e * Q, v);
    int rev = (e < E) ? (e + E) : (e - E);
    int d = ei[rev];
    float* ns = node_sum + (size_t)d * Q;
    #pragma unroll
    for (int q = 0; q < Q; ++q) atomicAdd(ns + q, v[q]);     // L2 f32 atomics
}

// ---------- kernel 2: hsum[q] = sum_n softmax(node_sum[n])[q] via async-LDS + WMMA ----
// Per wave: tile of 32 nodes.
//  (a) stage the contiguous 1KB node_sum slab into LDS with
//      global_load_async_to_lds_b128 (ASYNCcnt path; INST_OFFSET applies to both
//      LDS and global address, so offset:512 moves the second half).
//  (b) softmax each row from LDS, write marginals as bf16 into the B-operand tile.
//  (c) D = ones(16x32) x B(32x16) + C accumulates column sums on the matrix unit.
// B as packed here (contiguous k, not AMD's table order; A is all ones): lanes 0-15 hold K=0..15 (2 K per VGPR), lanes 16-31
// hold K=16..31; lane-within-group = column N.

__global__ void __launch_bounds__(256) k_node_h(const float* __restrict__ node_sum,
                                                float* __restrict__ hsum, int N) {
    __shared__ unsigned short smem[8][16 * 32];   // per-wave bf16 B tile   (8KB)
    __shared__ float          nsbuf[8][256];      // per-wave staged slab   (8KB)
    const int lane = threadIdx.x & 31;
    const int wave = threadIdx.x >> 5;
    const int gwave = blockIdx.x * (blockDim.x >> 5) + wave;
    const int nwaves = gridDim.x * (blockDim.x >> 5);
    const int ntiles = (N + 31) >> 5;
    unsigned short* t = smem[wave];

    // zero pad columns 8..15 once (never written again)
    #pragma unroll
    for (int j = 8; j < 16; ++j) t[j * 32 + lane] = 0;

    union { unsigned int u[8]; v16bf v; } A;      // all-ones bf16 matrix
    #pragma unroll
    for (int j = 0; j < 8; ++j) A.u[j] = 0x3F803F80u;

    v8f c = {0.f, 0.f, 0.f, 0.f, 0.f, 0.f, 0.f, 0.f};

    // LDS byte address of this lane's 16B chunk (flat-address low 32 bits == LDS offset)
    unsigned lds_dst = (unsigned)(uintptr_t)&nsbuf[wave][lane * 4];

    for (int tile = gwave; tile < ntiles; tile += nwaves) {   // wave-uniform trip count
        unsigned goff = (unsigned)tile * 1024u + (unsigned)lane * 16u;  // byte offset
        // WAR fence (prev iter LDS reads) -> async stage 1KB -> RAW fence
        asm volatile("s_wait_dscnt 0\n\t"
                     "global_load_async_to_lds_b128 %0, %1, %2\n\t"
                     "global_load_async_to_lds_b128 %0, %1, %2 offset:512\n\t"
                     "s_wait_asynccnt 0"
                     :: "v"(lds_dst), "v"(goff), "s"(node_sum) : "memory");

        int node = tile * 32 + lane;
        bool valid = node < N;
        float v[Q];
        {   // row of node (tile*32 + lane) sits at floats [lane*8 .. lane*8+8)
            const float4* lr = reinterpret_cast<const float4*>(&nsbuf[wave][lane * 8]);
            float4 a = lr[0], b2 = lr[1];
            v[0]=a.x; v[1]=a.y; v[2]=a.z; v[3]=a.w;
            v[4]=b2.x; v[5]=b2.y; v[6]=b2.z; v[7]=b2.w;
        }
        float m = v[0];
        #pragma unroll
        for (int q = 1; q < Q; ++q) m = fmaxf(m, v[q]);
        float s = 0.f;
        #pragma unroll
        for (int q = 0; q < Q; ++q) { v[q] = __expf(v[q] - m); s += v[q]; }
        float inv = valid ? (1.0f / s) : 0.0f;    // masked lanes contribute zeros
        #pragma unroll
        for (int q = 0; q < Q; ++q) t[q * 32 + lane] = f32_to_bf16(v[q] * inv);
        asm volatile("s_wait_dscnt 0" ::: "memory");          // wave-local store->load order
        union { unsigned int u[8]; v16bf v; } B;
        int n  = lane & 15;
        int k0 = (lane >> 4) << 4;
        const unsigned int* bp = reinterpret_cast<const unsigned int*>(&t[n * 32 + k0]);
        #pragma unroll
        for (int j = 0; j < 8; ++j) B.u[j] = bp[j];
        c = __builtin_amdgcn_wmma_f32_16x16x32_bf16(false, A.v, false, B.v,
                                                    (short)0, c, false, false);
    }
    // D rows are identical; lanes 0..15 hold column N=lane in VGPR0 -> q = lane for lane<8
    if (lane < Q) atomicAdd(&hsum[lane], c[0]);
}

// ---------- kernel 3: psi = softmax(node_sum[src] - log_term[rev] - h) ----------

__global__ void k_update(const int* __restrict__ ei, const float* __restrict__ node_sum,
                         const float* __restrict__ log_term, const float* __restrict__ hsum,
                         const float* __restrict__ beta, float* __restrict__ psi,
                         int E, int twoE, float invN) {
    int e = blockIdx.x * blockDim.x + threadIdx.x;
    if (e >= twoE) return;
    float hs = beta[0] * invN;
    int s   = ei[e];
    int rev = (e < E) ? (e + E) : (e - E);
    __builtin_prefetch(log_term + (size_t)(rev + 4096) * Q, 0, 0);
    float ns[Q], lt[Q];
    load_row8(node_sum + (size_t)s * Q, ns);
    load_row8(log_term + (size_t)rev * Q, lt);
    #pragma unroll
    for (int q = 0; q < Q; ++q) ns[q] = ns[q] - lt[q] - hs * hsum[q];
    softmax8(ns);
    store_row8(psi + (size_t)e * Q, ns);
}

// ---------- kernel 4: final marginal = softmax(node_sum) ----------

__global__ void k_marginal(const float* __restrict__ node_sum, float* __restrict__ marg, int N) {
    int n = blockIdx.x * blockDim.x + threadIdx.x;
    if (n >= N) return;
    float v[Q];
    load_row8(node_sum + (size_t)n * Q, v);
    softmax8(v);
    store_row8(marg + (size_t)n * Q, v);
}

// ---------- host ----------

extern "C" void kernel_launch(void* const* d_in, const int* in_sizes, int n_in,
                              void* d_out, int out_size, void* d_ws, size_t ws_size,
                              hipStream_t stream) {
    const int*   ei   = (const int*)d_in[0];     // [2][E] row-major: src_u | dst_u
    const float* psi0 = (const float*)d_in[1];   // [2E, Q]
    const float* beta = (const float*)d_in[2];   // [1]
    const int E    = in_sizes[0] / 2;
    const int twoE = 2 * E;
    const int N    = 50000;                      // fixed by setup_inputs

    float* out      = (float*)d_out;
    float* psi      = out;                           // message_map [2E*Q]
    float* marg     = out + (size_t)twoE * Q;        // marginal    [N*Q]
    float* node_sum = (float*)d_ws;                  // [N*Q]
    float* hsum     = node_sum + (size_t)N * Q;      // [Q] (contiguous with node_sum)
    float* log_term = node_sum + (size_t)N * Q + 512;// [2E*Q], 512B-aligned offset

    const int TB = 256;
    const int gEdge = (twoE + TB - 1) / TB;
    const int gNode = (((N + 31) / 32) + 7) / 8;     // 8 waves/block, 1 tile/wave
    const int gN    = (N + TB - 1) / TB;

    k_init_psi<<<gEdge, TB, 0, stream>>>(psi0, psi, twoE);

    for (int it = 0; it < NUM_ITER; ++it) {
        hipMemsetAsync(node_sum, 0, ((size_t)N * Q + Q) * sizeof(float), stream);
        k_scatter<<<gEdge, TB, 0, stream>>>(psi, ei, beta, log_term, node_sum, E, twoE);
        k_node_h<<<gNode, TB, 0, stream>>>(node_sum, hsum, N);
        k_update<<<gEdge, TB, 0, stream>>>(ei, node_sum, log_term, hsum, beta, psi,
                                           E, twoE, 1.0f / (float)N);
    }

    hipMemsetAsync(node_sum, 0, (size_t)N * Q * sizeof(float), stream);
    k_scatter<<<gEdge, TB, 0, stream>>>(psi, ei, beta, log_term, node_sum, E, twoE);
    k_marginal<<<gN, TB, 0, stream>>>(node_sum, marg, N);
}
